// RowAttentionWithPairBias_26242250179307
// MI455X (gfx1250) — matrix operand order repaired and checked by reading, not compiled, not run on hardware
//
#include <hip/hip_runtime.h>
#include <hip/hip_bf16.h>
#include <cstdint>

// Problem dims (fixed by the reference)
#define M_   128
#define I_   384
#define INC  256
#define PC   128
#define H_   8
#define C_   32
#define HC   256
#define MI   (M_ * I_)
#define LNEPS 1e-5f

typedef _Float16 f16;
typedef __attribute__((ext_vector_type(16))) _Float16 v16h;
typedef __attribute__((ext_vector_type(8)))  _Float16 v8h;
typedef __attribute__((ext_vector_type(8)))  float    v8f;
typedef __attribute__((ext_vector_type(4)))  unsigned int u32x4;
typedef __attribute__((ext_vector_type(8)))  int i32x8;
typedef __attribute__((ext_vector_type(4)))  int i32x4;

union V16 { v16h v; uint4 u4[2]; };

// ---------------- wave32 reductions ----------------
__device__ __forceinline__ float wave_sum32(float x){
  #pragma unroll
  for (int o = 16; o; o >>= 1) x += __shfl_xor(x, o, 32);
  return x;
}
// reduce within each 16-lane half of a wave32 (xor masks 8..1 stay in-half)
__device__ __forceinline__ float half_max16(float x){
  #pragma unroll
  for (int o = 8; o; o >>= 1) x = fmaxf(x, __shfl_xor(x, o, 32));
  return x;
}
__device__ __forceinline__ float half_sum16(float x){
  #pragma unroll
  for (int o = 8; o; o >>= 1) x += __shfl_xor(x, o, 32);
  return x;
}

// ---------------- WMMA helpers (CDNA5 16x16x32 f16 -> f32) ----------------
__device__ __forceinline__ v8f wmma_f16f32(v16h a, v16h b, v8f c){
  // 8 args: (neg_a, A, neg_b, B, c_mod, C, reuse_a, reuse_b)
  return __builtin_amdgcn_wmma_f32_16x16x32_f16(false, a, false, b, (short)0, c,
                                                false, false);
}

// A fragment (16x32 f16). Caller passes pointer to this lane's row (row = lane&15).
// Contiguous layout: elements e -> K = half*16+e (matches B fragment).
// Two 16-byte loads per lane. rowp must be 16B aligned.
__device__ __forceinline__ v16h load_a_frag(const f16* rowp, int halfid){
  V16 a;
  a.u4[0] = *(const uint4*)(rowp + halfid * 16);
  a.u4[1] = *(const uint4*)(rowp + halfid * 16 + 8);
  return a.v;
}
// B fragment (32x16 f16): lane col = lane&15, elements e -> K = half*16+e.
// One contiguous 32B load; p must be 32B aligned.
__device__ __forceinline__ v16h load_b_frag(const f16* p){
  return *(const v16h*)p;
}

// ---------------- optional TDM (Tensor Data Mover) staging ----------------
#if __has_builtin(__builtin_amdgcn_tensor_load_to_lds) && __has_builtin(__builtin_amdgcn_s_wait_tensorcnt)
#define ATHENA_HAS_TDM 1
// 2D f16 tile load: rows x width elements, row stride `stride` elements, into LDS.
// This toolchain exposes the 6-arg builtin:
//   (uint32x4 g0, int32x8 g1, int32x4 g2, int32x4 g3, int32x8 extra, i32 cpol)
__device__ __forceinline__ void tdm_load_2d_f16(unsigned lds_off, const void* gsrc,
                                                unsigned width, unsigned rows,
                                                unsigned stride){
  unsigned long long ga = (unsigned long long)(uintptr_t)gsrc;
  u32x4 g0; i32x8 g1; i32x4 g2; i32x4 g3; i32x8 g4;
  g0[0] = 1u;                                              // count=1 valid, no gather
  g0[1] = lds_off;                                         // lds_addr (bytes)
  g0[2] = (unsigned)(ga & 0xffffffffull);                  // global_addr[31:0]
  g0[3] = (unsigned)((ga >> 32) & 0x1ffffffull) | (2u<<30);// global_addr[56:32] | type=2
  g1[0] = (int)(1u << 16);                                 // data_size=1 (2 bytes)
  g1[1] = (int)((width & 0xffffu) << 16);                  // tensor_dim0[15:0] @ [63:48]
  g1[2] = (int)((width >> 16) | ((rows & 0xffffu) << 16)); // dim0 hi | tensor_dim1 lo
  g1[3] = (int)((rows >> 16) | ((width & 0xffffu) << 16)); // dim1 hi | tile_dim0
  g1[4] = (int)(rows & 0xffffu);                           // tile_dim1 (tile_dim2=0)
  g1[5] = (int)stride;                                     // tensor_dim0_stride[31:0]
  g1[6] = 0; g1[7] = 0;
  g2[0]=0; g2[1]=0; g2[2]=0; g2[3]=0;
  g3[0]=0; g3[1]=0; g3[2]=0; g3[3]=0;
  g4[0]=0; g4[1]=0; g4[2]=0; g4[3]=0; g4[4]=0; g4[5]=0; g4[6]=0; g4[7]=0;
  __builtin_amdgcn_tensor_load_to_lds(g0, g1, g2, g3, g4, 0);
}
#endif

// ================= K0: weight prep (f32 -> f16, transposed to [n][k]) ============
__global__ __launch_bounds__(256) void prep_w_kernel(
    const float* __restrict__ Wq, const float* __restrict__ Wk,
    const float* __restrict__ Wv, const float* __restrict__ Wg,
    const float* __restrict__ Wf, f16* __restrict__ wt4, f16* __restrict__ wtf){
  const int mat = blockIdx.y;          // 0..4
  const float* src = (mat==0)?Wq:(mat==1)?Wk:(mat==2)?Wv:(mat==3)?Wg:Wf;
  f16* dst = (mat < 4) ? (wt4 + (size_t)mat * INC * INC) : wtf;
  const int k = blockIdx.x;            // source row (coalesced read)
  const int n = threadIdx.x;           // source col
  dst[(size_t)n * INC + k] = (f16)src[(size_t)k * INC + n];
}

// ================= K1: LayerNorm(x1d) -> f16 =====================================
__global__ __launch_bounds__(256) void ln1_kernel(
    const float* __restrict__ x, const float* __restrict__ g,
    const float* __restrict__ b, f16* __restrict__ y){
  const int lane = threadIdx.x & 31;
  const int wave = threadIdx.x >> 5;
  const size_t row = (size_t)blockIdx.x * 8 + wave;   // row of [MI, 256]
  const float* xr = x + row * INC + lane * 8;
  float v[8];
  float4 p0 = *(const float4*)(xr);
  float4 p1 = *(const float4*)(xr + 4);
  v[0]=p0.x; v[1]=p0.y; v[2]=p0.z; v[3]=p0.w;
  v[4]=p1.x; v[5]=p1.y; v[6]=p1.z; v[7]=p1.w;
  float s = 0.f, q = 0.f;
  #pragma unroll
  for (int e = 0; e < 8; ++e){ s += v[e]; q += v[e]*v[e]; }
  s = wave_sum32(s); q = wave_sum32(q);
  const float mean = s * (1.f/INC);
  const float var  = q * (1.f/INC) - mean*mean;
  const float rstd = rsqrtf(var + LNEPS);
  v8h o;
  #pragma unroll
  for (int e = 0; e < 8; ++e){
    const int ch = lane*8 + e;
    o[e] = (f16)((v[e]-mean)*rstd*g[ch] + b[ch]);
  }
  *(v8h*)(y + row * INC + lane*8) = o;
}

// ================= K2: LayerNorm(x2d) + pair bias ================================
// pb[h][i][j] = sum_p LN(x2d[i,j,:])[p] * W2d[p][h]
__global__ __launch_bounds__(256) void pair_kernel(
    const float* __restrict__ x2d, const float* __restrict__ g,
    const float* __restrict__ b, const float* __restrict__ W2d,
    float* __restrict__ pb){
  __shared__ float w2s[PC * H_];
  for (int t = threadIdx.x; t < PC*H_; t += 256) w2s[t] = W2d[t];
  __syncthreads();
  const int lane = threadIdx.x & 31;
  const int wave = threadIdx.x >> 5;
  const size_t gid = (size_t)blockIdx.x * 8 + wave;   // i*384 + j
  const int i = (int)(gid / I_), j = (int)(gid % I_);
  const float* xr = x2d + gid * PC;
  float v[4];
  #pragma unroll
  for (int e = 0; e < 4; ++e) v[e] = xr[lane + 32*e];
  float s = 0.f, q = 0.f;
  #pragma unroll
  for (int e = 0; e < 4; ++e){ s += v[e]; q += v[e]*v[e]; }
  s = wave_sum32(s); q = wave_sum32(q);
  const float mean = s * (1.f/PC);
  const float rstd = rsqrtf(q * (1.f/PC) - mean*mean + LNEPS);
  float xn[4];
  #pragma unroll
  for (int e = 0; e < 4; ++e){
    const int ch = lane + 32*e;
    xn[e] = (v[e]-mean)*rstd*g[ch] + b[ch];
  }
  #pragma unroll
  for (int hh = 0; hh < H_; ++hh){
    float a = 0.f;
    #pragma unroll
    for (int e = 0; e < 4; ++e) a += xn[e] * w2s[(lane + 32*e)*H_ + hh];
    a = wave_sum32(a);
    if (lane == 0) pb[((size_t)hh * I_ + i) * I_ + j] = a;
  }
}

// ================= K3: fused Q/K/V/Gate projection (WMMA GEMM) ===================
// y = x1h[MI,256] @ {Wq|Wk|Wv|Wg}; q scaled by 1/sqrt(C); gate -> sigmoid.
__global__ __launch_bounds__(128) void qkvg_kernel(
    const f16* __restrict__ x1h, const f16* __restrict__ wt4,
    const float* __restrict__ bg,
    f16* __restrict__ qb, f16* __restrict__ kb, f16* __restrict__ vb,
    f16* __restrict__ gate){
  const int tid = threadIdx.x, lane = tid & 31, wave = tid >> 5;
  const int halfid = lane >> 4, l16 = lane & 15;
  const int row0  = blockIdx.x * 16;
  const int ncol0 = blockIdx.y * 64 + wave * 16;   // 0..1023
  const int mat   = ncol0 >> 8;                    // 0=q 1=k 2=v 3=gate
  const int nloc  = ncol0 & 255;
  const f16* wt = wt4 + (size_t)mat * INC * INC;   // [n][k] f16
  const f16* ap = x1h + (size_t)(row0 + l16) * INC;
  const f16* bp = wt  + (size_t)(nloc + l16) * INC;
  v8f acc;
  #pragma unroll
  for (int r = 0; r < 8; ++r) acc[r] = 0.f;
  #pragma unroll
  for (int kc = 0; kc < INC; kc += 32){
    v16h a = load_a_frag(ap + kc, halfid);
    v16h b = load_b_frag(bp + kc + halfid * 16);
    acc = wmma_f16f32(a, b, acc);
  }
  const int col = nloc + l16;
  #pragma unroll
  for (int r = 0; r < 8; ++r){
    const int R = row0 + r + 8*halfid;             // 0..MI-1
    if (mat == 3){
      float gv = acc[r] + bg[col];
      gv = 1.f / (1.f + __expf(-gv));
      gate[(size_t)R * HC + col] = (f16)gv;
    } else {
      const int m = R / I_, i = R % I_;
      const int hh = col >> 5, cc = col & 31;
      float v = acc[r];
      if (mat == 0) v *= 0.17677669529663689f;     // 1/sqrt(32) folded into q
      f16* dst = (mat == 0) ? qb : (mat == 1) ? kb : vb;
      dst[(((size_t)(m*H_ + hh)) * I_ + i) * C_ + cc] = (f16)v;
    }
  }
}

// ================= K4: flash attention per (m, h, 32-row block) ==================
// Per wave: 16x16 O tile (ihalf = wave>>1, ct = wave&1). Online softmax over J.
__global__ __launch_bounds__(128) void attn_kernel(
    const f16* __restrict__ qb, const f16* __restrict__ kb,
    const f16* __restrict__ vb, const f16* __restrict__ gateb,
    const float* __restrict__ pb, const float* __restrict__ mask,
    f16* __restrict__ attb){
  __shared__ __align__(32) f16 vT[C_][400];        // V transposed [c][j], padded
  __shared__ __align__(16) f16 Pw[4][16][40];      // per-wave P bounce tile, padded
#ifdef ATHENA_HAS_TDM
  __shared__ __align__(16) f16 vstage[I_ * C_];    // linear V staged by TDM
#endif
  const int tid = threadIdx.x, lane = tid & 31, wave = tid >> 5;
  const int halfid = lane >> 4, l16 = lane & 15;
  const int i0 = blockIdx.x * 32;
  const int h  = blockIdx.y;
  const int m  = blockIdx.z;
  const f16* qh = qb + ((size_t)(m*H_ + h)) * I_ * C_;
  const f16* kh = kb + ((size_t)(m*H_ + h)) * I_ * C_;
  const f16* vh = vb + ((size_t)(m*H_ + h)) * I_ * C_;
  const float* pbh   = pb + (size_t)h * I_ * I_;
  const float* maskm = mask + (size_t)m * I_;

  // ---- stage V^T in LDS (B-operand then reads 32B contiguous per lane)
#ifdef ATHENA_HAS_TDM
  if (wave == 0){
    tdm_load_2d_f16((unsigned)(uintptr_t)&vstage[0], vh, C_, I_, C_);
    __builtin_amdgcn_s_wait_tensorcnt(0);
  }
  __syncthreads();
  const f16* vsrc = vstage;
#else
  const f16* vsrc = vh;
#endif
  for (int j = tid; j < I_; j += 128){
    const f16* src = vsrc + (size_t)j * C_;
    #pragma unroll
    for (int c = 0; c < C_; ++c) vT[c][j] = src[c];
  }
  __syncthreads();

  const int ihalf = wave >> 1;                      // which 16 rows of the 32
  const int ct    = wave & 1;                       // which 16 of C=32 outputs
  const int arow  = i0 + ihalf*16 + l16;
  const v16h aq = load_a_frag(qh + (size_t)arow * C_, halfid);  // K = C = 32
  f16* pbase = &Pw[wave][0][0];

  v8f oacc;
  float Mrow[8], Lrow[8];
  #pragma unroll
  for (int r = 0; r < 8; ++r){ oacc[r] = 0.f; Mrow[r] = -1e30f; Lrow[r] = 0.f; }

  for (int jc = 0; jc < I_ / 32; ++jc){
    const int j0 = jc * 32;
    // ---- S = q.k^T for two 16-col tiles (K^T B-frag = contiguous k row bytes)
    v16h bk0 = load_b_frag(kh + (size_t)(j0      + l16) * C_ + halfid*16);
    v16h bk1 = load_b_frag(kh + (size_t)(j0 + 16 + l16) * C_ + halfid*16);
    v8f z;
    #pragma unroll
    for (int r = 0; r < 8; ++r) z[r] = 0.f;
    v8f s0 = wmma_f16f32(aq, bk0, z);
    v8f s1 = wmma_f16f32(aq, bk1, z);
    const int jA = j0 + l16, jB = j0 + 16 + l16;
    const float mbA = 10000.f * (maskm[jA] - 1.f);
    const float mbB = 10000.f * (maskm[jB] - 1.f);
    float nm[8];
    #pragma unroll
    for (int r = 0; r < 8; ++r){
      const int ig = i0 + ihalf*16 + r + 8*halfid;
      s0[r] += pbh[(size_t)ig * I_ + jA] + mbA;
      s1[r] += pbh[(size_t)ig * I_ + jB] + mbB;
      float cm = half_max16(fmaxf(s0[r], s1[r]));
      nm[r] = fmaxf(Mrow[r], cm);
    }
    #pragma unroll
    for (int r = 0; r < 8; ++r){
      const float alpha = __expf(Mrow[r] - nm[r]);
      const float p0 = __expf(s0[r] - nm[r]);
      const float p1 = __expf(s1[r] - nm[r]);
      const float rs = half_sum16(p0 + p1);
      Lrow[r] = Lrow[r] * alpha + rs;
      Mrow[r] = nm[r];
      oacc[r] *= alpha;
      const int rloc = r + 8*halfid;                // C-layout row -> LDS row
      pbase[rloc*40 + l16]      = (f16)p0;
      pbase[rloc*40 + 16 + l16] = (f16)p1;
    }
    // DS ops are in-order per wave: A-frag reload below sees the stores above.
    V16 ap;
    const f16* prow = pbase + l16 * 40;             // A row = lane&15
    ap.u4[0] = *(const uint4*)(prow + halfid*16);
    ap.u4[1] = *(const uint4*)(prow + halfid*16 + 8);
    const v16h bv = load_b_frag(&vT[ct*16 + l16][j0 + halfid*16]);
    oacc = wmma_f16f32(ap.v, bv, oacc);
  }
  // ---- epilogue: 1/L, gate, write attention output [MI, H*C] f16
  #pragma unroll
  for (int r = 0; r < 8; ++r){
    const int ig = i0 + ihalf*16 + r + 8*halfid;
    const int c  = ct*16 + l16;
    const size_t idx = ((size_t)(m*I_ + ig)) * HC + h*C_ + c;
    float val = oacc[r] / Lrow[r];
    val *= (float)gateb[idx];
    attb[idx] = (f16)val;
  }
}

// ================= K5: output projection (WMMA GEMM) + bf -> f32 =================
__global__ __launch_bounds__(128) void outproj_kernel(
    const f16* __restrict__ attb, const f16* __restrict__ wtf,
    const float* __restrict__ bf, float* __restrict__ out){
  const int tid = threadIdx.x, lane = tid & 31, wave = tid >> 5;
  const int halfid = lane >> 4, l16 = lane & 15;
  const int row0  = blockIdx.x * 16;
  const int ncol0 = blockIdx.y * 64 + wave * 16;
  const f16* ap = attb + (size_t)(row0 + l16) * HC;
  const f16* bp = wtf  + (size_t)(ncol0 + l16) * HC;
  v8f acc;
  #pragma unroll
  for (int r = 0; r < 8; ++r) acc[r] = 0.f;
  #pragma unroll
  for (int kc = 0; kc < HC; kc += 32){
    v16h a = load_a_frag(ap + kc, halfid);
    v16h b = load_b_frag(bp + kc + halfid * 16);
    acc = wmma_f16f32(a, b, acc);
  }
  const int col = ncol0 + l16;
  const float bias = bf[col];
  #pragma unroll
  for (int r = 0; r < 8; ++r){
    const int R = row0 + r + 8*halfid;
    out[(size_t)R * INC + col] = acc[r] + bias;
  }
}

// ================================ launch =========================================
extern "C" void kernel_launch(void* const* d_in, const int* in_sizes, int n_in,
                              void* d_out, int out_size, void* d_ws, size_t ws_size,
                              hipStream_t stream){
  (void)in_sizes; (void)n_in; (void)out_size; (void)ws_size;
  const float* x1d   = (const float*)d_in[0];
  const float* x2d   = (const float*)d_in[1];
  const float* mask  = (const float*)d_in[2];
  const float* ln1_g = (const float*)d_in[3];
  const float* ln1_b = (const float*)d_in[4];
  const float* ln2_g = (const float*)d_in[5];
  const float* ln2_b = (const float*)d_in[6];
  const float* Wq    = (const float*)d_in[7];
  const float* Wk    = (const float*)d_in[8];
  const float* Wv    = (const float*)d_in[9];
  const float* W2d   = (const float*)d_in[10];
  const float* Wg    = (const float*)d_in[11];
  const float* bg    = (const float*)d_in[12];
  const float* Wf    = (const float*)d_in[13];
  const float* bf    = (const float*)d_in[14];
  float* out = (float*)d_out;

  // workspace carve (256B aligned)
  char* ws = (char*)d_ws;
  size_t off = 0;
  auto carve = [&](size_t bytes) -> char* {
    char* p = ws + off;
    off += (bytes + 255) & ~(size_t)255;
    return p;
  };
  f16*   x1h  = (f16*)  carve((size_t)MI * INC * 2);
  f16*   wt4  = (f16*)  carve((size_t)4 * INC * INC * 2);
  f16*   wtf  = (f16*)  carve((size_t)INC * INC * 2);
  f16*   qb   = (f16*)  carve((size_t)M_ * H_ * I_ * C_ * 2);
  f16*   kb   = (f16*)  carve((size_t)M_ * H_ * I_ * C_ * 2);
  f16*   vb   = (f16*)  carve((size_t)M_ * H_ * I_ * C_ * 2);
  f16*   gate = (f16*)  carve((size_t)MI * HC * 2);
  float* pbuf = (float*)carve((size_t)H_ * I_ * I_ * 4);
  f16*   attb = (f16*)  carve((size_t)MI * HC * 2);

  hipLaunchKernelGGL(prep_w_kernel, dim3(INC, 5), dim3(256), 0, stream,
                     Wq, Wk, Wv, Wg, Wf, wt4, wtf);
  hipLaunchKernelGGL(ln1_kernel, dim3(MI / 8), dim3(256), 0, stream,
                     x1d, ln1_g, ln1_b, x1h);
  hipLaunchKernelGGL(pair_kernel, dim3((I_ * I_) / 8), dim3(256), 0, stream,
                     x2d, ln2_g, ln2_b, W2d, pbuf);
  hipLaunchKernelGGL(qkvg_kernel, dim3(MI / 16, 16), dim3(128), 0, stream,
                     x1h, wt4, bg, qb, kb, vb, gate);
  hipLaunchKernelGGL(attn_kernel, dim3(I_ / 32, H_, M_), dim3(128), 0, stream,
                     qb, kb, vb, gate, pbuf, mask, attb);
  hipLaunchKernelGGL(outproj_kernel, dim3(MI / 16, 4), dim3(128), 0, stream,
                     attb, wtf, bf, out);
}
